// MultiheadAttention_23553600651376
// MI455X (gfx1250) — compile-verified, matrix operand order checked by reading, not run on hardware
//
#include <hip/hip_runtime.h>
#include <hip/hip_bf16.h>

typedef __attribute__((ext_vector_type(16))) _Float16 v16h;
typedef __attribute__((ext_vector_type(8)))  float    v8f;
typedef __attribute__((ext_vector_type(4)))  unsigned int u32x4;
typedef __attribute__((ext_vector_type(4)))  float    f32x4;

#define Bb 4
#define Tt 2048
#define Ee 1024
#define Hh 16
#define Dd 64
#define BHTD (Bb * Hh * Tt * Dd)   // 8388608

union Frag { v16h v; u32x4 u[2]; };
union H8   { u32x4 u; _Float16 h[8]; };
union H4   { unsigned long long u; _Float16 h[4]; };

// Async global -> LDS copy, 16 bytes per lane (CDNA5; tracked by ASYNCcnt).
// lds = wave-relative LDS byte address (low 32 bits of generic shared ptr),
// g   = per-lane 64-bit global address.
__device__ __forceinline__ void async_g2l_b128(unsigned lds, const _Float16* g) {
  asm volatile("global_load_async_to_lds_b128 %0, %1, off"
               :: "v"(lds), "v"(g)
               : "memory");
}
__device__ __forceinline__ void wait_async0() {
  asm volatile("s_wait_asynccnt 0x0" ::: "memory");
}

// ---------------------------------------------------------------------------
// fp32 -> f16 conversion (vectorized, grid-stride)
// ---------------------------------------------------------------------------
__global__ __launch_bounds__(256) void cvt_f32_f16(const float* __restrict__ s,
                                                   _Float16* __restrict__ d,
                                                   int n4) {
  int i = blockIdx.x * blockDim.x + threadIdx.x;
  int st = gridDim.x * blockDim.x;
  for (; i < n4; i += st) {
    f32x4 v = ((const f32x4*)s)[i];
    H4 o;
    o.h[0] = (_Float16)v.x; o.h[1] = (_Float16)v.y;
    o.h[2] = (_Float16)v.z; o.h[3] = (_Float16)v.w;
    ((unsigned long long*)d)[i] = o.u;
  }
}

// ---------------------------------------------------------------------------
// WMMA GEMM: C[M,N] = A[M,K] * Bw[N,K]^T  (f16 in, f32 accumulate)
// Block tile 128x128, K-step 32, 8 waves; each wave -> 32x64 (2x4 frags).
// Double-buffered LDS fed by async global->LDS copies (software pipeline).
// mode 0: scatter-epilogue writing f16 QKV in (3,B,H,T,D), Q scaled 1/sqrt(D)
// mode 1: plain fp32 store to outf[M,N]
// ---------------------------------------------------------------------------
__global__ __launch_bounds__(256) void gemm_wmma(
    const _Float16* __restrict__ A,
    const _Float16* __restrict__ Bw,
    _Float16* __restrict__ qkv,
    float* __restrict__ outf,
    int N, int K, int mode) {
  __shared__ _Float16 As[2 * 128 * 40];
  __shared__ _Float16 Bs[2 * 128 * 40];

  const int tid   = threadIdx.x;
  const int lane  = tid & 31;
  const int wave  = tid >> 5;
  const int kh    = lane >> 4;   // K-half (ISA A/B frag layout)
  const int ln    = lane & 15;
  const int m0    = blockIdx.x * 128;
  const int n0    = blockIdx.y * 128;
  const int waveM = (wave & 3) * 32;
  const int waveN = (wave >> 2) * 64;

  const unsigned asBase = (unsigned)(uintptr_t)(&As[0]);
  const unsigned bsBase = (unsigned)(uintptr_t)(&Bs[0]);

  auto stage = [&](int k0, int pb) {
#pragma unroll
    for (int i = 0; i < 2; ++i) {
      int c = tid + i * 256;
      int row = c >> 2, cc = c & 3;           // 128 rows x 4 chunks of 8 f16
      unsigned off = (unsigned)(pb * 5120 + row * 40 + cc * 8) * 2u;
      async_g2l_b128(asBase + off, A  + (size_t)(m0 + row) * K + k0 + cc * 8);
      async_g2l_b128(bsBase + off, Bw + (size_t)(n0 + row) * K + k0 + cc * 8);
    }
  };

  v8f acc[2][4];
#pragma unroll
  for (int i = 0; i < 2; ++i)
#pragma unroll
    for (int j = 0; j < 4; ++j)
#pragma unroll
      for (int r = 0; r < 8; ++r) acc[i][j][r] = 0.0f;

  const int nk = K / 32;
  int buf = 0;
  stage(0, 0);
  for (int it = 0; it < nk; ++it) {
    wait_async0();        // this wave's async tiles (cur buffer) have landed
    __syncthreads();      // all waves: cur tile ready, prev reads retired
    if (it + 1 < nk) stage((it + 1) * 32, buf ^ 1);

    const _Float16* Ab = &As[buf * 5120];
    const _Float16* Bbs = &Bs[buf * 5120];
    Frag af[2], bf[4];
#pragma unroll
    for (int mf = 0; mf < 2; ++mf) {
      const _Float16* p = &Ab[(waveM + mf * 16 + ln) * 40 + kh * 8];
      af[mf].u[0] = *(const u32x4*)p;
      af[mf].u[1] = *(const u32x4*)(p + 16);
    }
#pragma unroll
    for (int nf = 0; nf < 4; ++nf) {
      const _Float16* p = &Bbs[(waveN + nf * 16 + ln) * 40 + kh * 8];
      bf[nf].u[0] = *(const u32x4*)p;
      bf[nf].u[1] = *(const u32x4*)(p + 16);
    }
#pragma unroll
    for (int mf = 0; mf < 2; ++mf)
#pragma unroll
      for (int nf = 0; nf < 4; ++nf)
        acc[mf][nf] = __builtin_amdgcn_wmma_f32_16x16x32_f16(
            false, af[mf].v, false, bf[nf].v, (short)0, acc[mf][nf], false, false);
    buf ^= 1;
  }

  if (mode == 0) {
    const float qscale = 0.125f;  // 1/sqrt(64)
#pragma unroll
    for (int mf = 0; mf < 2; ++mf)
#pragma unroll
      for (int nf = 0; nf < 4; ++nf)
#pragma unroll
        for (int r = 0; r < 8; ++r) {
          int m = m0 + waveM + mf * 16 + r + kh * 8;
          int f = n0 + waveN + nf * 16 + ln;
          int s = f >> 10, h = (f >> 6) & 15, d = f & 63;
          int b = m >> 11, t = m & 2047;
          float val = acc[mf][nf][r];
          if (s == 0) val *= qscale;
          qkv[(size_t)s * BHTD +
              (((size_t)(b * Hh + h) * Tt + t) * Dd + d)] = (_Float16)val;
        }
  } else {
#pragma unroll
    for (int mf = 0; mf < 2; ++mf)
#pragma unroll
      for (int nf = 0; nf < 4; ++nf)
#pragma unroll
        for (int r = 0; r < 8; ++r) {
          int m = m0 + waveM + mf * 16 + r + kh * 8;
          int f = n0 + waveN + nf * 16 + ln;
          outf[(size_t)m * N + f] = acc[mf][nf][r];
        }
  }
}

// ---------------------------------------------------------------------------
// Flash attention (causal, online softmax). 256 threads, 128 q-rows/block.
// Q,K,V: f16 (B*H, T, D).  Output: f16 (B, T, H*D) for the final projection.
// K tile staged with async global->LDS; V transposed through registers.
// ---------------------------------------------------------------------------
__global__ __launch_bounds__(256) void attn_wmma(
    const _Float16* __restrict__ Qg,
    const _Float16* __restrict__ Kg,
    const _Float16* __restrict__ Vg,
    _Float16* __restrict__ Og) {
  __shared__ _Float16 Ks[64 * 72];        // key-major K tile
  __shared__ _Float16 Vts[64 * 72];       // d-major (transposed) V tile
  __shared__ _Float16 Ps[8 * 16 * 72];    // per-wave P tile (16 x 64)

  const int tid  = threadIdx.x;
  const int lane = tid & 31;
  const int wave = tid >> 5;
  const int kh   = lane >> 4;
  const int ln   = lane & 15;
  const int nqb  = Tt / 128;              // 16
  const int qb   = blockIdx.x % nqb;
  const int bh   = blockIdx.x / nqb;
  const int qbase = qb * 128 + wave * 16; // this wave's 16 q-rows

  const _Float16* Qh = Qg + (size_t)bh * Tt * Dd;
  const _Float16* Kh = Kg + (size_t)bh * Tt * Dd;
  const _Float16* Vh = Vg + (size_t)bh * Tt * Dd;

  const unsigned ksBase = (unsigned)(uintptr_t)(&Ks[0]);

  // Q fragments for this wave's 16 rows (K-dim 64 -> 2 frags)
  Frag qf[2];
#pragma unroll
  for (int kq = 0; kq < 2; ++kq) {
    const _Float16* p = Qh + (size_t)(qbase + ln) * Dd + kq * 32 + kh * 8;
    qf[kq].u[0] = *(const u32x4*)p;
    qf[kq].u[1] = *(const u32x4*)(p + 16);
  }

  v8f o[4];
  float mr[8], lr[8];
#pragma unroll
  for (int df = 0; df < 4; ++df)
#pragma unroll
    for (int r = 0; r < 8; ++r) o[df][r] = 0.0f;
#pragma unroll
  for (int r = 0; r < 8; ++r) { mr[r] = -1e30f; lr[r] = 0.0f; }

  const int nkb = qb * 2 + 2;             // causal: key blocks of 64
  for (int kb = 0; kb < nkb; ++kb) {
    // stage K tile via async global->LDS (rows = keys)
#pragma unroll
    for (int i = 0; i < 2; ++i) {
      int c = tid + i * 256;
      int row = c >> 3, cc = c & 7;
      async_g2l_b128(ksBase + (unsigned)(row * 72 + cc * 8) * 2u,
                     Kh + (size_t)(kb * 64 + row) * Dd + cc * 8);
    }
    // stage V^T tile (rows = d) through registers
#pragma unroll
    for (int i = 0; i < 2; ++i) {
      int c = tid + i * 256;
      int row = c >> 3, cc = c & 7;
      H8 hv;
      hv.u = *(const u32x4*)(Vh + (size_t)(kb * 64 + row) * Dd + cc * 8);
#pragma unroll
      for (int j = 0; j < 8; ++j) Vts[(cc * 8 + j) * 72 + row] = hv.h[j];
      if (kb + 1 < nkb)
        __builtin_prefetch(Vh + (size_t)((kb + 1) * 64 + row) * Dd + cc * 8, 0, 0);
    }
    wait_async0();
    __syncthreads();

    const bool active = (kb * 64) <= (qbase + 15);
    if (active) {
      // S = Q * K^T  (16 x 64)
      v8f s[4];
#pragma unroll
      for (int nf = 0; nf < 4; ++nf) {
#pragma unroll
        for (int r = 0; r < 8; ++r) s[nf][r] = 0.0f;
#pragma unroll
        for (int kq = 0; kq < 2; ++kq) {
          Frag bfr;
          const _Float16* p = &Ks[(nf * 16 + ln) * 72 + kq * 32 + kh * 8];
          bfr.u[0] = *(const u32x4*)p;
          bfr.u[1] = *(const u32x4*)(p + 16);
          s[nf] = __builtin_amdgcn_wmma_f32_16x16x32_f16(
              false, qf[kq].v, false, bfr.v, (short)0, s[nf], false, false);
        }
        // causal mask (only diagonal-ish frags need it)
        int kgmax = kb * 64 + nf * 16 + 15;
        if (kgmax > qbase) {
          int kg = kb * 64 + nf * 16 + ln;
#pragma unroll
          for (int r = 0; r < 8; ++r) {
            int qg = qbase + r + kh * 8;
            s[nf][r] = (kg > qg) ? -1e30f : s[nf][r];
          }
        }
      }

      // online softmax: row max / rescale / exp / row sum
      float rowm[8], alpha[8], rsum[8];
#pragma unroll
      for (int r = 0; r < 8; ++r) {
        float v = s[0][r];
        v = fmaxf(v, s[1][r]); v = fmaxf(v, s[2][r]); v = fmaxf(v, s[3][r]);
#pragma unroll
        for (int off = 1; off < 16; off <<= 1)
          v = fmaxf(v, __shfl_xor(v, off, 32));
        rowm[r]  = fmaxf(mr[r], v);
        alpha[r] = __expf(mr[r] - rowm[r]);
        rsum[r]  = 0.0f;
      }

      _Float16* Pw = &Ps[wave * 16 * 72];
#pragma unroll
      for (int nf = 0; nf < 4; ++nf)
#pragma unroll
        for (int r = 0; r < 8; ++r) {
          float p = __expf(s[nf][r] - rowm[r]);
          rsum[r] += p;
          Pw[(r + kh * 8) * 72 + nf * 16 + ln] = (_Float16)p;
        }
#pragma unroll
      for (int r = 0; r < 8; ++r) {
        float t = rsum[r];
#pragma unroll
        for (int off = 1; off < 16; off <<= 1)
          t += __shfl_xor(t, off, 32);
        lr[r] = lr[r] * alpha[r] + t;
        mr[r] = rowm[r];
      }
#pragma unroll
      for (int df = 0; df < 4; ++df)
#pragma unroll
        for (int r = 0; r < 8; ++r) o[df][r] *= alpha[r];

      // O += P * V (P re-read from LDS in A-frag layout; same-wave LDS is in-order)
      Frag pf[2];
#pragma unroll
      for (int kq = 0; kq < 2; ++kq) {
        const _Float16* p = &Pw[ln * 72 + kq * 32 + kh * 8];
        pf[kq].u[0] = *(const u32x4*)p;
        pf[kq].u[1] = *(const u32x4*)(p + 16);
      }
#pragma unroll
      for (int df = 0; df < 4; ++df)
#pragma unroll
        for (int kq = 0; kq < 2; ++kq) {
          Frag bfr;
          const _Float16* p = &Vts[(df * 16 + ln) * 72 + kq * 32 + kh * 8];
          bfr.u[0] = *(const u32x4*)p;
          bfr.u[1] = *(const u32x4*)(p + 16);
          o[df] = __builtin_amdgcn_wmma_f32_16x16x32_f16(
              false, pf[kq].v, false, bfr.v, (short)0, o[df], false, false);
        }
    }
    __syncthreads();
  }

  // finalize: divide by l, store f16 in (B, T, H*D)
  const int b = bh / Hh, h = bh % Hh;
#pragma unroll
  for (int r = 0; r < 8; ++r) {
    float inv = 1.0f / lr[r];
    int tq = qbase + r + kh * 8;
#pragma unroll
    for (int df = 0; df < 4; ++df) {
      Og[((size_t)b * Tt + tq) * Ee + h * Dd + df * 16 + ln] =
          (_Float16)(o[df][r] * inv);
    }
  }
}

// ---------------------------------------------------------------------------
extern "C" void kernel_launch(void* const* d_in, const int* in_sizes, int n_in,
                              void* d_out, int out_size, void* d_ws, size_t ws_size,
                              hipStream_t stream) {
  const float* q32    = (const float*)d_in[0];
  const float* wqkv32 = (const float*)d_in[3];
  const float* wout32 = (const float*)d_in[4];
  float* out = (float*)d_out;

  // workspace layout (f16): X, Wqkv, Wout, QKV (3x BHTD), O_attn
  _Float16* Xh   = (_Float16*)d_ws;            // 8,388,608
  _Float16* Wqkv = Xh + (size_t)BHTD;          // 3,145,728
  _Float16* Wout = Wqkv + 3145728;             // 1,048,576
  _Float16* QKVb = Wout + 1048576;             // 3 * 8,388,608
  _Float16* Oatt = QKVb + (size_t)3 * BHTD;    // 8,388,608

  cvt_f32_f16<<<4096, 256, 0, stream>>>(q32,    Xh,   BHTD / 4);
  cvt_f32_f16<<<3072, 256, 0, stream>>>(wqkv32, Wqkv, 3145728 / 4);
  cvt_f32_f16<<<1024, 256, 0, stream>>>(wout32, Wout, 1048576 / 4);

  // QKV projection: M=8192, N=3072, K=1024
  gemm_wmma<<<dim3(64, 24), 256, 0, stream>>>(Xh, Wqkv, QKVb, nullptr,
                                              3 * Ee, Ee, 0);
  // attention: (B*H) * (T/128) blocks
  attn_wmma<<<dim3(Bb * Hh * (Tt / 128)), 256, 0, stream>>>(
      QKVb, QKVb + (size_t)BHTD, QKVb + (size_t)2 * BHTD, Oatt);
  // output projection: M=8192, N=1024, K=1024
  gemm_wmma<<<dim3(64, 8), 256, 0, stream>>>(Oatt, Wout, nullptr, out,
                                             Ee, Ee, 1);
}
